// Multihead_64201171141389
// MI455X (gfx1250) — matrix operand order repaired and checked by reading, not compiled, not run on hardware
//
#include <hip/hip_runtime.h>
#include <hip/hip_bf16.h>

typedef __attribute__((ext_vector_type(16))) __bf16 v16bf;
typedef __attribute__((ext_vector_type(8)))  float  v8f;

#define BB 4
#define SS 2048
#define EE 1024
#define HH 16
#define HS 64

__device__ __forceinline__ v8f wmma_bf16(v16bf a, v16bf b, v8f c) {
  return __builtin_amdgcn_wmma_f32_16x16x32_bf16(false, a, false, b, (short)0, c, false, false);
}

// Build a v16bf fragment from two 16-byte runs (ds_load_b128 / global_load_b128).
__device__ __forceinline__ v16bf load_frag_pair(const __bf16* p0, const __bf16* p1) {
  union { v16bf v; uint4 u[2]; } f;
  f.u[0] = *(const uint4*)p0;
  f.u[1] = *(const uint4*)p1;
  return f.v;
}

// A-fragment (16x32 bf16) from row-major [M][K] source, ld = K stride (elements).
// ISA 7.12.2: lane m = lane%16, half = lane/16; elem i -> K = (i/8)*16 + half*8 + (i%8)
// => two contiguous 8-element runs at K = h8 and K = 16 + h8.
__device__ __forceinline__ v16bf load_a_frag(const __bf16* buf, int ld) {
  int lane = threadIdx.x & 31;
  int m = lane & 15, h8 = (lane >> 4) << 3;
  const __bf16* r = buf + m * ld + h8;
  return load_frag_pair(r, r + 16);
}

// B-fragment (32x16) from a TRANSPOSED source buf[n][k] (k contiguous, ld = k stride).
// Layout: lane n = lane%16; elem i -> K = (i/8)*16 + (lane/16)*8 + (i%8)  => runs at h8 and 16 + h8.
__device__ __forceinline__ v16bf load_bt_frag(const __bf16* buf, int ld) {
  int lane = threadIdx.x & 31;
  int n = lane & 15, h8 = (lane >> 4) << 3;
  const __bf16* r = buf + n * ld + h8;
  return load_frag_pair(r, r + 16);
}

__device__ __forceinline__ float half_reduce_max(float x) {
  x = fmaxf(x, __shfl_xor(x, 1, 32));
  x = fmaxf(x, __shfl_xor(x, 2, 32));
  x = fmaxf(x, __shfl_xor(x, 4, 32));
  x = fmaxf(x, __shfl_xor(x, 8, 32));
  return x;
}
__device__ __forceinline__ float half_reduce_sum(float x) {
  x += __shfl_xor(x, 1, 32);
  x += __shfl_xor(x, 2, 32);
  x += __shfl_xor(x, 4, 32);
  x += __shfl_xor(x, 8, 32);
  return x;
}

// pack 4 floats -> 4 bf16 (8 bytes)
__device__ __forceinline__ uint2 pack4_bf16(float4 f) {
  union { __bf16 h[4]; uint2 u; } p;
  p.h[0] = (__bf16)f.x; p.h[1] = (__bf16)f.y;
  p.h[2] = (__bf16)f.z; p.h[3] = (__bf16)f.w;
  return p.u;
}

// ---------------- Kernel 1: QKV projection ----------------
// grid (64, 16, 3), block 256. Tile: 128 rows of x  x  64 cols (one head), K=1024.
// q,k: bf16 [B,H,S,64].  v: bf16 [B,H,64,S] (transposed for the PV stage).
__global__ __launch_bounds__(256) void qkv_gemm_kernel(
    const float* __restrict__ x, const float* __restrict__ Wq,
    const float* __restrict__ Wk, const float* __restrict__ Wv,
    __bf16* __restrict__ q, __bf16* __restrict__ k, __bf16* __restrict__ v) {
  __shared__ __attribute__((aligned(16))) __bf16 lds_a[128 * 32];   // [m][k]
  __shared__ __attribute__((aligned(16))) __bf16 lds_bt[64 * 32];   // [n][k] transposed

  const int mt = blockIdx.x, h = blockIdx.y, sel = blockIdx.z;
  const float* W = (sel == 0 ? Wq : sel == 1 ? Wk : Wv) + (size_t)h * EE * HS;
  __bf16* outp = (sel == 0 ? q : sel == 1 ? k : v);

  const int row0 = mt * 128;
  const int t = threadIdx.x, wave = t >> 5, lane = t & 31;
  const int half = lane >> 4, ln = lane & 15;

  v8f acc[4];
#pragma unroll
  for (int i = 0; i < 4; ++i) acc[i] = (v8f){0.f, 0.f, 0.f, 0.f, 0.f, 0.f, 0.f, 0.f};

  for (int kc = 0; kc < EE; kc += 32) {
    // stage A tile 128x32: float4 global loads -> packed bf16x4 LDS stores
#pragma unroll
    for (int it = 0; it < 4; ++it) {
      int base = (t + it * 256) * 4;           // element index, 4096 total
      int r = base >> 5, c = base & 31;
      float4 xa = *(const float4*)&x[(size_t)(row0 + r) * EE + kc + c];
      *(uint2*)&lds_a[base] = pack4_bf16(xa);
    }
    // stage B tile 32x64 transposed into [n][k]: coalesced float4 reads, b16 scatter
#pragma unroll
    for (int it = 0; it < 2; ++it) {
      int base = (t + it * 256) * 4;           // idx = kk*64 + n, 2048 total
      int kk = base >> 6, n = base & 63;
      float4 w = *(const float4*)&W[(size_t)(kc + kk) * HS + n];
      lds_bt[(n + 0) * 32 + kk] = (__bf16)w.x;
      lds_bt[(n + 1) * 32 + kk] = (__bf16)w.y;
      lds_bt[(n + 2) * 32 + kk] = (__bf16)w.z;
      lds_bt[(n + 3) * 32 + kk] = (__bf16)w.w;
    }
    if (kc + 32 < EE)
      __builtin_prefetch(&x[(size_t)(row0 + (t >> 5)) * EE + kc + 32], 0, 1);
    __syncthreads();

    v16bf a = load_a_frag(&lds_a[wave * 16 * 32], 32);
#pragma unroll
    for (int nt = 0; nt < 4; ++nt) {
      v16bf bf = load_bt_frag(&lds_bt[nt * 16 * 32], 32);
      acc[nt] = wmma_bf16(a, bf, acc[nt]);
    }
    __syncthreads();
  }

  // store
#pragma unroll
  for (int nt = 0; nt < 4; ++nt)
#pragma unroll
    for (int j = 0; j < 8; ++j) {
      int grow = row0 + wave * 16 + j + 8 * half;     // index into B*S
      int bb = grow >> 11, s = grow & (SS - 1);
      int d = nt * 16 + ln;
      __bf16 val = (__bf16)acc[nt][j];
      if (sel == 2)  // V transposed: [B,H,64,S]
        outp[(((size_t)bb * HH + h) * HS + d) * SS + s] = val;
      else           // Q,K: [B,H,S,64]
        outp[(((size_t)bb * HH + h) * SS + s) * HS + d] = val;
    }
}

// ---------------- Kernel 2: causal flash attention ----------------
// grid (16, 16, 4) = (S/128, H, B), block 256. Wave owns 16 query rows.
__global__ __launch_bounds__(256) void flash_attn_kernel(
    const __bf16* __restrict__ q, const __bf16* __restrict__ k,
    const __bf16* __restrict__ v, __bf16* __restrict__ ao) {
  __shared__ __attribute__((aligned(16))) __bf16 lds_k[32 * 64];    // [key][d]
  __shared__ __attribute__((aligned(16))) __bf16 lds_vt[64 * 32];   // [d][key]
  __shared__ __attribute__((aligned(16))) __bf16 lds_p[8][16 * 32]; // per-wave P

  const int qt = blockIdx.x, h = blockIdx.y, b = blockIdx.z;
  const int t = threadIdx.x, wave = t >> 5, lane = t & 31;
  const int half = lane >> 4, ln = lane & 15;
  const int h8 = half << 3;

  const size_t bh = ((size_t)b * HH + h) * SS * HS;
  const __bf16* qp = q + bh;
  const __bf16* kp = k + bh;
  const __bf16* vpt = v + bh;   // [64][S] transposed layout

  const int q0 = qt * 128 + wave * 16;

  // Q fragments (16x64 -> two A-frags) via global_load_b128 pairs
  const __bf16* qbase = qp + (size_t)(q0 + ln) * HS;
  v16bf aq0 = load_frag_pair(qbase + h8, qbase + h8 + 16);
  v16bf aq1 = load_frag_pair(qbase + 32 + h8, qbase + 48 + h8);

  v8f accv[4];
#pragma unroll
  for (int i = 0; i < 4; ++i) accv[i] = (v8f){0.f, 0.f, 0.f, 0.f, 0.f, 0.f, 0.f, 0.f};
  float mrow[8], lrow[8];
#pragma unroll
  for (int j = 0; j < 8; ++j) { mrow[j] = -__builtin_inff(); lrow[j] = 0.f; }

  const int nkb = qt * 4 + 4;   // causal bound, uniform over workgroup
  for (int kb = 0; kb < nkb; ++kb) {
    __syncthreads();  // previous iteration's LDS consumers done
    {
      // K tile: 32 keys x 64 dims, contiguous copy (256 x uint4)
      ((uint4*)lds_k)[t] = ((const uint4*)(kp + (size_t)kb * 32 * HS))[t];
      // V tile: [d][key] from transposed workspace; thread t -> d = t/4, 8 keys
      int d = t >> 2, c8 = (t & 3) * 8;
      *(uint4*)&lds_vt[d * 32 + c8] =
          *(const uint4*)&vpt[(size_t)d * SS + kb * 32 + c8];
      if (kb + 1 < nkb) __builtin_prefetch(kp + (size_t)(kb + 1) * 32 * HS + t * 8, 0, 1);
    }
    __syncthreads();

    // scores: two 16x16 tiles; lds_k[key][d] is exactly the transposed B source
    v8f s0 = (v8f){0.f, 0.f, 0.f, 0.f, 0.f, 0.f, 0.f, 0.f};
    v8f s1 = (v8f){0.f, 0.f, 0.f, 0.f, 0.f, 0.f, 0.f, 0.f};
    s0 = wmma_bf16(aq0, load_bt_frag(&lds_k[0 * 16 * 64 + 0], 64), s0);
    s0 = wmma_bf16(aq1, load_bt_frag(&lds_k[0 * 16 * 64 + 32], 64), s0);
    s1 = wmma_bf16(aq0, load_bt_frag(&lds_k[1 * 16 * 64 + 0], 64), s1);
    s1 = wmma_bf16(aq1, load_bt_frag(&lds_k[1 * 16 * 64 + 32], 64), s1);

    // online softmax update per owned row
#pragma unroll
    for (int j = 0; j < 8; ++j) {
      int row = q0 + j + 8 * half;
      int key0 = kb * 32 + ln;
      float x0 = s0[j] * 0.125f;
      float x1 = s1[j] * 0.125f;
      if (key0 > row) x0 = -1e30f;
      if (key0 + 16 > row) x1 = -1e30f;
      float rm = half_reduce_max(fmaxf(x0, x1));
      float mnew = fmaxf(mrow[j], rm);
      float p0 = __expf(x0 - mnew);
      float p1 = __expf(x1 - mnew);
      float rs = half_reduce_sum(p0 + p1);
      float sc = __expf(mrow[j] - mnew);
      lrow[j] = lrow[j] * sc + rs;
      mrow[j] = mnew;
#pragma unroll
      for (int nt = 0; nt < 4; ++nt) accv[nt][j] *= sc;
      int m16 = (j + 8 * half) * 32;
      lds_p[wave][m16 + ln] = (__bf16)p0;
      lds_p[wave][m16 + 16 + ln] = (__bf16)p1;
    }
    __syncthreads();  // P staging visible (uniform barrier)

    // out += P(16x32) x V(32x64); lds_vt[d][key] is the transposed B source
    v16bf pa = load_a_frag(lds_p[wave], 32);
#pragma unroll
    for (int nt = 0; nt < 4; ++nt) {
      v16bf bv = load_bt_frag(&lds_vt[nt * 16 * 32], 32);
      accv[nt] = wmma_bf16(pa, bv, accv[nt]);
    }
  }

  // epilogue: normalize, store head-major concat [B*S, H*hs] bf16
#pragma unroll
  for (int nt = 0; nt < 4; ++nt)
#pragma unroll
    for (int j = 0; j < 8; ++j) {
      int row = q0 + j + 8 * half;
      float o = accv[nt][j] / lrow[j];
      ao[((size_t)b * SS + row) * (HH * HS) + h * HS + nt * 16 + ln] = (__bf16)o;
    }
}

// ---------------- Kernel 3: output projection ----------------
// grid (64, 16), block 256. [8192,1024]bf16 x [1024,1024]f32 -> f32 out.
__global__ __launch_bounds__(256) void out_proj_kernel(
    const __bf16* __restrict__ ao, const float* __restrict__ Wc,
    float* __restrict__ out) {
  __shared__ __attribute__((aligned(16))) __bf16 lds_a[128 * 32];   // [m][k]
  __shared__ __attribute__((aligned(16))) __bf16 lds_bt[64 * 32];   // [n][k]

  const int row0 = blockIdx.x * 128, col0 = blockIdx.y * 64;
  const int t = threadIdx.x, wave = t >> 5, lane = t & 31;
  const int half = lane >> 4, ln = lane & 15;
  const int N = HH * HS;  // 1024

  v8f acc[4];
#pragma unroll
  for (int i = 0; i < 4; ++i) acc[i] = (v8f){0.f, 0.f, 0.f, 0.f, 0.f, 0.f, 0.f, 0.f};

  for (int kc = 0; kc < N; kc += 32) {
    // A tile: bf16 -> bf16, 128-bit copies (512 x uint4)
#pragma unroll
    for (int it = 0; it < 2; ++it) {
      int base = (t + it * 256) * 8;           // element index, 4096 total
      int r = base >> 5, c = base & 31;
      *(uint4*)&lds_a[base] = *(const uint4*)&ao[(size_t)(row0 + r) * N + kc + c];
    }
    // B tile transposed: coalesced float4 reads of Wc, b16 scatter
#pragma unroll
    for (int it = 0; it < 2; ++it) {
      int base = (t + it * 256) * 4;           // idx = kk*64 + n
      int kk = base >> 6, n = base & 63;
      float4 w = *(const float4*)&Wc[(size_t)(kc + kk) * EE + col0 + n];
      lds_bt[(n + 0) * 32 + kk] = (__bf16)w.x;
      lds_bt[(n + 1) * 32 + kk] = (__bf16)w.y;
      lds_bt[(n + 2) * 32 + kk] = (__bf16)w.z;
      lds_bt[(n + 3) * 32 + kk] = (__bf16)w.w;
    }
    if (kc + 32 < N)
      __builtin_prefetch(&ao[(size_t)(row0 + (t >> 5)) * N + kc + 32], 0, 1);
    __syncthreads();

    v16bf a = load_a_frag(&lds_a[wave * 16 * 32], 32);
#pragma unroll
    for (int nt = 0; nt < 4; ++nt) {
      v16bf bf = load_bt_frag(&lds_bt[nt * 16 * 32], 32);
      acc[nt] = wmma_bf16(a, bf, acc[nt]);
    }
    __syncthreads();
  }

#pragma unroll
  for (int nt = 0; nt < 4; ++nt)
#pragma unroll
    for (int j = 0; j < 8; ++j)
      out[(size_t)(row0 + wave * 16 + j + 8 * half) * EE + col0 + nt * 16 + ln] =
          acc[nt][j];
}

extern "C" void kernel_launch(void* const* d_in, const int* in_sizes, int n_in,
                              void* d_out, int out_size, void* d_ws, size_t ws_size,
                              hipStream_t stream) {
  const float* x  = (const float*)d_in[0];
  const float* Wq = (const float*)d_in[1];
  const float* Wk = (const float*)d_in[2];
  const float* Wv = (const float*)d_in[3];
  const float* Wc = (const float*)d_in[4];
  float* out = (float*)d_out;

  const size_t qkv_elems = (size_t)BB * HH * SS * HS;  // 8,388,608
  __bf16* q  = (__bf16*)d_ws;
  __bf16* k  = q + qkv_elems;
  __bf16* v  = k + qkv_elems;          // transposed [B,H,64,S]
  __bf16* ao = v + qkv_elems;          // [B*S, H*hs] bf16

  qkv_gemm_kernel<<<dim3(BB * SS / 128, HH, 3), 256, 0, stream>>>(x, Wq, Wk, Wv, q, k, v);
  flash_attn_kernel<<<dim3(SS / 128, HH, BB), 256, 0, stream>>>(q, k, v, ao);
  out_proj_kernel<<<dim3(BB * SS / 128, EE / 64), 256, 0, stream>>>(ao, Wc, out);
}
